// CondTransportKernel_79053168050794
// MI455X (gfx1250) — matrix operand order repaired and checked by reading, not compiled, not run on hardware
//
#include <hip/hip_runtime.h>
#include <stdint.h>

typedef __bf16 bf16;
typedef __attribute__((ext_vector_type(16))) bf16  v16bf;
typedef __attribute__((ext_vector_type(8)))  bf16  v8bf;
typedef __attribute__((ext_vector_type(8)))  float v8f;
typedef __attribute__((ext_vector_type(4)))  unsigned v4u;
typedef __attribute__((ext_vector_type(8)))  int v8i;
typedef __attribute__((ext_vector_type(4)))  int v4i;
typedef __attribute__((address_space(3)))   char lds_char;

#define NX_    8192
#define NQ_    8192
#define DMEAN  64
#define DVAR   96
#define DY_    32
#define KSPLIT 8     // lambda_gemm k-slices  -> 64*8  = 512 blocks
#define NSPLIT 4     // rbf_gemm   n-slices  -> 64*4  = 256 blocks per path

// ---------------------------------------------------------------------------
// helpers
// ---------------------------------------------------------------------------
static __device__ __forceinline__ v16bf join8(v8bf lo, v8bf hi) {
  return __builtin_shufflevector(lo, hi, 0,1,2,3,4,5,6,7,8,9,10,11,12,13,14,15);
}
static __device__ __forceinline__ v16bf frag16(const bf16* p0, const bf16* p1) {
  return join8(*(const v8bf*)p0, *(const v8bf*)p1);
}
static __device__ __forceinline__ v8f wmma_bf16(v16bf a, v16bf b, v8f c) {
  return __builtin_amdgcn_wmma_f32_16x16x32_bf16(false, a, false, b, (short)0, c,
                                                 false, false);
}
static __device__ __forceinline__ unsigned lds_offset(void* p) {
  return (unsigned)(unsigned long long)(lds_char*)p;
}

// Tensor Data Mover: 2D tile (tile_dim0=32 f32 cols, tile_dim1=128 rows) from a
// row-major tensor with dim0_stride=8192, DMA'd into LDS with 16B padding after
// every 32 dwords (row stride becomes 36 dwords = bank-conflict-free layout).
static __device__ __forceinline__ void tdm_load_tile(unsigned long long gaddr,
                                                     unsigned lds_byte_off) {
  const v4u g0 = { 1u,                                   // count=1, user mode
                   lds_byte_off,                          // lds_addr
                   (unsigned)gaddr,                       // global_addr[31:0]
                   (unsigned)((gaddr >> 32) & 0x01FFFFFFull) | (2u << 30) }; // type=2
  const v8i g1 = { (int)((2u << 16)      // data_size = 4B
                       | (1u << 20)      // pad_enable
                       | (4u << 22)      // pad_interval: 32 dwords
                       | (3u << 25)),    // pad_amount:   4 dwords (16B)
                   (int)(0x2000u << 16), // tensor_dim0 = 8192 (low 16 in [31:16])
                   (int)(0x2000u << 16), // dim0 hi=0 | tensor_dim1 = 8192 low16
                   (int)(32u << 16),     // dim1 hi=0 | tile_dim0 = 32
                   128,                  // tile_dim1 = 128, tile_dim2 = 0
                   8192,                 // tensor_dim0_stride (low 32)
                   0, 0 };
  const v4i gz4 = { 0, 0, 0, 0 };
  const v8i gz8 = { 0, 0, 0, 0, 0, 0, 0, 0 };
  __builtin_amdgcn_tensor_load_to_lds(g0, g1, gz4, gz4, gz8, 0);
}

// ---------------------------------------------------------------------------
// prep: concatenated query matrices (bf16) + row norms (f32)
// ---------------------------------------------------------------------------
__global__ __launch_bounds__(256) void prep_queries(
    const float* __restrict__ x_mu, const float* __restrict__ y_eta,
    const float* __restrict__ y_mean, const float* __restrict__ y_var,
    bf16* __restrict__ Qm, bf16* __restrict__ Qv,
    float* __restrict__ Qmn, float* __restrict__ Qvn)
{
  const int wave = threadIdx.x >> 5, lane = threadIdx.x & 31;
  const int row  = blockIdx.x * 8 + wave;
  const float m = x_mu  [(size_t)row * 32 + lane];
  const float s = y_mean[(size_t)row * 32 + lane] + y_var[(size_t)row * 32 + lane];
  const float e = 0.01f * y_eta[(size_t)(NQ_ - 1 - row) * 32 + lane];
  Qm[(size_t)row * DMEAN + lane]      = (bf16)m;
  Qm[(size_t)row * DMEAN + 32 + lane] = (bf16)s;
  Qv[(size_t)row * DVAR  + lane]      = (bf16)m;
  Qv[(size_t)row * DVAR  + 32 + lane] = (bf16)e;
  Qv[(size_t)row * DVAR  + 64 + lane] = (bf16)s;
  float nm = m * m + s * s;
  float nv = nm + e * e;
#pragma unroll
  for (int off = 16; off > 0; off >>= 1) {
    nm += __shfl_xor(nm, off, 32);
    nv += __shfl_xor(nv, off, 32);
  }
  if (lane == 0) { Qmn[row] = nm; Qvn[row] = nv; }
}

__global__ __launch_bounds__(256) void prep_X(
    const float* __restrict__ X, bf16* __restrict__ Xb, float* __restrict__ Xn, int D)
{
  const int wave = threadIdx.x >> 5, lane = threadIdx.x & 31;
  const int row  = blockIdx.x * 8 + wave;
  float s = 0.f;
  for (int c = lane; c < D; c += 32) {
    float v = X[(size_t)row * D + c];
    Xb[(size_t)row * D + c] = (bf16)v;
    s += v * v;
  }
#pragma unroll
  for (int off = 16; off > 0; off >>= 1) s += __shfl_xor(s, off, 32);
  if (lane == 0) Xn[row] = s;
}

// ---------------------------------------------------------------------------
// Phase 1 (split-K): Part[ks] = Kinv[:, ks-slice] @ Z[ks-slice, :]
// Kinv tiles streamed by the Tensor Data Mover into double-buffered LDS (f32),
// converted to bf16 at fragment build (v_cvt_pk_bf16_f32).
// ---------------------------------------------------------------------------
__global__ __launch_bounds__(256) void lambda_gemm(
    const float* __restrict__ Kinv, const float* __restrict__ Z,
    float* __restrict__ Part)
{
  __shared__ __align__(16) float At[2][128][36];  // TDM-padded f32 tiles
  __shared__ __align__(16) bf16  ZT[32][40];      // Z^T tile [d][k]
  const int t = threadIdx.x;
  const int wave = t >> 5, lane = t & 31;
  const int lh = lane >> 4, ln = lane & 15;
  const int rowbase = blockIdx.x * 128;
  const int kb0 = blockIdx.y * (NX_ / KSPLIT);
  const int ITERS = (NX_ / KSPLIT) / 32;

  const unsigned long long gbase =
      (unsigned long long)(uintptr_t)Kinv +
      ((unsigned long long)rowbase * NX_ + (unsigned long long)kb0) * 4ull;

  const int zk = (t * 4) >> 5, zd = (t * 4) & 31;
  float4 zreg = *(const float4*)(Z + (size_t)(kb0 + zk) * DY_ + zd);
  if (wave == 0) tdm_load_tile(gbase, lds_offset(&At[0][0][0]));

  v8f acc0 = {}, acc1 = {};
  for (int it = 0; it < ITERS; ++it) {
    if (wave == 0) __builtin_amdgcn_s_wait_tensorcnt(0);
    __syncthreads();                               // tile `it` visible; LDS free
    ZT[zd + 0][zk] = (bf16)zreg.x; ZT[zd + 1][zk] = (bf16)zreg.y;
    ZT[zd + 2][zk] = (bf16)zreg.z; ZT[zd + 3][zk] = (bf16)zreg.w;
    if (it + 1 < ITERS) {
      zreg = *(const float4*)(Z + (size_t)(kb0 + (it + 1) * 32 + zk) * DY_ + zd);
      if (wave == 0)
        tdm_load_tile(gbase + (unsigned long long)(it + 1) * 128ull,
                      lds_offset(&At[(it + 1) & 1][0][0]));
    }
    __syncthreads();

    const float* ar = &At[it & 1][wave * 16 + ln][lh * 16];
    const float4 f0 = *(const float4*)(ar);
    const float4 f1 = *(const float4*)(ar + 4);
    const float4 f2 = *(const float4*)(ar + 8);
    const float4 f3 = *(const float4*)(ar + 12);
    v16bf A;
    A[0] = (bf16)f0.x; A[1] = (bf16)f0.y; A[2]  = (bf16)f0.z; A[3]  = (bf16)f0.w;
    A[4] = (bf16)f1.x; A[5] = (bf16)f1.y; A[6]  = (bf16)f1.z; A[7]  = (bf16)f1.w;
    A[8] = (bf16)f2.x; A[9] = (bf16)f2.y; A[10] = (bf16)f2.z; A[11] = (bf16)f2.w;
    A[12] = (bf16)f3.x; A[13] = (bf16)f3.y; A[14] = (bf16)f3.z; A[15] = (bf16)f3.w;
    const bf16* b0 = &ZT[ln][lh * 16];
    const bf16* b1 = &ZT[16 + ln][lh * 16];
    acc0 = wmma_bf16(A, frag16(b0, b0 + 8), acc0);
    acc1 = wmma_bf16(A, frag16(b1, b1 + 8), acc1);
  }
  float* out0 = Part +
      ((size_t)blockIdx.y * NX_ + rowbase + wave * 16 + 8 * lh) * DY_ + ln;
#pragma unroll
  for (int r = 0; r < 8; ++r) {
    out0[(size_t)r * DY_]      = acc0[r];
    out0[(size_t)r * DY_ + 16] = acc1[r];
  }
}

__global__ __launch_bounds__(256) void reduce_lambda(
    const float* __restrict__ P, float* __restrict__ O)
{
  const int i = blockIdx.x * 256 + threadIdx.x;
  float4 s = ((const float4*)P)[i];
#pragma unroll
  for (int p = 1; p < KSPLIT; ++p) {
    const float4 v = ((const float4*)(P + (size_t)p * NX_ * DY_))[i];
    s.x += v.x; s.y += v.y; s.z += v.z; s.w += v.w;
  }
  ((float4*)O)[i] = s;
}

// ---------------------------------------------------------------------------
// Phase 2 (fused, split-N): Zpart[ns][q,d] = sum_{n in slice} exp(..) Lam[n,d]
// Register-double-buffered staging so global loads overlap the WMMA chain.
// ---------------------------------------------------------------------------
template <int D>
__global__ __launch_bounds__(256) void rbf_gemm(
    const bf16* __restrict__ Xb, const float* __restrict__ Xn,
    const bf16* __restrict__ Qb, const float* __restrict__ Qn,
    const float* __restrict__ Lam, float* __restrict__ Zpart, float negScale)
{
  constexpr int SX  = D + 8;
  constexpr int TOT = 32 * (D / 8);
  constexpr int CH  = (TOT + 255) / 256;
  __shared__ __align__(16) bf16 Qs[128][SX];
  __shared__ __align__(16) bf16 Xs[32][SX];
  __shared__ __align__(16) bf16 LT[32][40];
  __shared__ __align__(16) bf16 Ks[8][16][40];
  __shared__ float qns[128];
  __shared__ float xns[32];

  const int t = threadIdx.x;
  const int wave = t >> 5, lane = t & 31;
  const int lh = lane >> 4, ln = lane & 15;
  const int qbase = blockIdx.x * 128;
  const int nb0 = blockIdx.y * (NX_ / NSPLIT);
  const int ITERS = (NX_ / NSPLIT) / 32;

  for (int c = t; c < 128 * (D / 8); c += 256) {
    const int r = c / (D / 8), off = (c % (D / 8)) * 8;
    *(uint4*)&Qs[r][off] = *(const uint4*)(Qb + (size_t)(qbase + r) * D + off);
  }
  if (t < 128) qns[t] = Qn[qbase + t];
  __syncthreads();
  float qnr[8];
#pragma unroll
  for (int r = 0; r < 8; ++r) qnr[r] = qns[wave * 16 + 8 * lh + r];

  const int zk = (t * 4) >> 5, zd = (t * 4) & 31;
  uint4 xreg[CH];
  float4 lreg;
  float xnreg = 0.f;
  auto preload = [&](int nb) {
#pragma unroll
    for (int j = 0; j < CH; ++j) {
      const int c = t + j * 256;
      if (c < TOT) {
        const int r = c / (D / 8), off = (c % (D / 8)) * 8;
        xreg[j] = *(const uint4*)(Xb + (size_t)(nb + r) * D + off);
      }
    }
    lreg = *(const float4*)(Lam + (size_t)(nb + zk) * DY_ + zd);
    if (t < 32) xnreg = Xn[nb + t];
  };
  preload(nb0);

  v8f acc0 = {}, acc1 = {};
  for (int it = 0; it < ITERS; ++it) {
    __syncthreads();
#pragma unroll
    for (int j = 0; j < CH; ++j) {
      const int c = t + j * 256;
      if (c < TOT) {
        const int r = c / (D / 8), off = (c % (D / 8)) * 8;
        *(uint4*)&Xs[r][off] = xreg[j];
      }
    }
    LT[zd + 0][zk] = (bf16)lreg.x; LT[zd + 1][zk] = (bf16)lreg.y;
    LT[zd + 2][zk] = (bf16)lreg.z; LT[zd + 3][zk] = (bf16)lreg.w;
    if (t < 32) xns[t] = xnreg;
    if (it + 1 < ITERS) preload(nb0 + (it + 1) * 32);
    __syncthreads();

    v8f S0 = {}, S1 = {};
#pragma unroll
    for (int kc = 0; kc < D; kc += 32) {
      const bf16* ar = &Qs[wave * 16 + ln][kc + lh * 16];
      const v16bf A = frag16(ar, ar + 8);
      const bf16* b0 = &Xs[ln][kc + lh * 16];
      const bf16* b1 = &Xs[16 + ln][kc + lh * 16];
      S0 = wmma_bf16(A, frag16(b0, b0 + 8), S0);
      S1 = wmma_bf16(A, frag16(b1, b1 + 8), S1);
    }
    const float nn0 = xns[ln], nn1 = xns[16 + ln];
#pragma unroll
    for (int r = 0; r < 8; ++r) {
      const int M = r + 8 * lh;
      const float d20 = fmaxf(qnr[r] + nn0 - 2.0f * S0[r], 0.0f);
      const float d21 = fmaxf(qnr[r] + nn1 - 2.0f * S1[r], 0.0f);
      Ks[wave][M][ln]      = (bf16)__builtin_amdgcn_exp2f(d20 * negScale);
      Ks[wave][M][16 + ln] = (bf16)__builtin_amdgcn_exp2f(d21 * negScale);
    }
    const bf16* kr = &Ks[wave][ln][lh * 16];
    const v16bf KA = frag16(kr, kr + 8);
    const bf16* lb0 = &LT[ln][lh * 16];
    const bf16* lb1 = &LT[16 + ln][lh * 16];
    acc0 = wmma_bf16(KA, frag16(lb0, lb0 + 8), acc0);
    acc1 = wmma_bf16(KA, frag16(lb1, lb1 + 8), acc1);
  }
  float* out0 = Zpart +
      ((size_t)blockIdx.y * NQ_ + qbase + wave * 16 + 8 * lh) * DY_ + ln;
#pragma unroll
  for (int r = 0; r < 8; ++r) {
    out0[(size_t)r * DY_]      = acc0[r];
    out0[(size_t)r * DY_ + 16] = acc1[r];
  }
}

// out = y_mean + y_var + sum_p zm[p] + sum_p zv[p]   (deterministic order)
__global__ __launch_bounds__(256) void combine_out(
    const float* __restrict__ ym, const float* __restrict__ yv,
    const float* __restrict__ zm, const float* __restrict__ zv,
    float* __restrict__ out)
{
  const int i = blockIdx.x * 256 + threadIdx.x;
  const float4 a = ((const float4*)ym)[i];
  const float4 b = ((const float4*)yv)[i];
  float4 s;
  s.x = a.x + b.x; s.y = a.y + b.y; s.z = a.z + b.z; s.w = a.w + b.w;
#pragma unroll
  for (int p = 0; p < NSPLIT; ++p) {
    const float4 u = ((const float4*)(zm + (size_t)p * NQ_ * DY_))[i];
    const float4 v = ((const float4*)(zv + (size_t)p * NQ_ * DY_))[i];
    s.x += u.x + v.x; s.y += u.y + v.y; s.z += u.z + v.z; s.w += u.w + v.w;
  }
  ((float4*)out)[i] = s;
}

// ---------------------------------------------------------------------------
extern "C" void kernel_launch(void* const* d_in, const int* in_sizes, int n_in,
                              void* d_out, int out_size, void* d_ws, size_t ws_size,
                              hipStream_t stream) {
  (void)in_sizes; (void)n_in; (void)out_size; (void)ws_size;
  const float* x_mu   = (const float*)d_in[0];
  const float* y_eta  = (const float*)d_in[1];
  const float* y_mean = (const float*)d_in[2];
  const float* y_var  = (const float*)d_in[3];
  const float* X_mean = (const float*)d_in[4];
  const float* X_var  = (const float*)d_in[5];
  const float* Z_mean = (const float*)d_in[6];
  const float* Z_var  = (const float*)d_in[7];
  const float* Kinv_m = (const float*)d_in[8];
  const float* Kinv_v = (const float*)d_in[9];
  float* out = (float*)d_out;

  uintptr_t p = (uintptr_t)d_ws;
  auto carve = [&](size_t bytes) -> void* {
    uintptr_t r = (p + 255) & ~(uintptr_t)255;
    p = r + bytes;
    return (void*)r;
  };
  float* LamPM = (float*)carve((size_t)KSPLIT * NX_ * DY_ * 4);
  float* LamPV = (float*)carve((size_t)KSPLIT * NX_ * DY_ * 4);
  float* LamM  = (float*)carve((size_t)NX_ * DY_ * 4);
  float* LamV  = (float*)carve((size_t)NX_ * DY_ * 4);
  float* ZPm   = (float*)carve((size_t)NSPLIT * NQ_ * DY_ * 4);
  float* ZPv   = (float*)carve((size_t)NSPLIT * NQ_ * DY_ * 4);
  bf16*  XmB   = (bf16*) carve((size_t)NX_ * DMEAN * 2);
  bf16*  XvB   = (bf16*) carve((size_t)NX_ * DVAR  * 2);
  bf16*  QmB   = (bf16*) carve((size_t)NQ_ * DMEAN * 2);
  bf16*  QvB   = (bf16*) carve((size_t)NQ_ * DVAR  * 2);
  float* XmN   = (float*)carve((size_t)NX_ * 4);
  float* XvN   = (float*)carve((size_t)NX_ * 4);
  float* QmN   = (float*)carve((size_t)NQ_ * 4);
  float* QvN   = (float*)carve((size_t)NQ_ * 4);

  const float negScale = -1.0f / (2.0f * 8.0f * 8.0f * 0.6931471805599453f);

  prep_queries<<<NQ_ / 8, 256, 0, stream>>>(x_mu, y_eta, y_mean, y_var,
                                            QmB, QvB, QmN, QvN);
  prep_X<<<NX_ / 8, 256, 0, stream>>>(X_mean, XmB, XmN, DMEAN);
  prep_X<<<NX_ / 8, 256, 0, stream>>>(X_var,  XvB, XvN, DVAR);
  lambda_gemm<<<dim3(NX_ / 128, KSPLIT), 256, 0, stream>>>(Kinv_m, Z_mean, LamPM);
  lambda_gemm<<<dim3(NX_ / 128, KSPLIT), 256, 0, stream>>>(Kinv_v, Z_var,  LamPV);
  reduce_lambda<<<(NX_ * DY_ / 4) / 256, 256, 0, stream>>>(LamPM, LamM);
  reduce_lambda<<<(NX_ * DY_ / 4) / 256, 256, 0, stream>>>(LamPV, LamV);
  rbf_gemm<DMEAN><<<dim3(NQ_ / 128, NSPLIT), 256, 0, stream>>>(
      XmB, XmN, QmB, QmN, LamM, ZPm, negScale);
  rbf_gemm<DVAR ><<<dim3(NQ_ / 128, NSPLIT), 256, 0, stream>>>(
      XvB, XvN, QvB, QvN, LamV, ZPv, negScale);
  combine_out<<<(NQ_ * DY_ / 4) / 256, 256, 0, stream>>>(y_mean, y_var, ZPm, ZPv, out);
}
